// SingleHeadSiameseAttention_73332271612117
// MI455X (gfx1250) — matrix operand order repaired and checked by reading, not compiled, not run on hardware
//
#include <hip/hip_runtime.h>

// ---------------- types ----------------
typedef __bf16 bf16;
typedef __attribute__((ext_vector_type(16))) __bf16 v16bf;
typedef __attribute__((ext_vector_type(8)))  float  v8f;

union F8      { v8f  v; float e[8];  };
union BF16x16 { v16bf v; bf16 e[16]; };

__device__ __forceinline__ bf16 to_bf16(float f) {
  union { float f; unsigned u; } a; a.f = f;
  unsigned r = (a.u + 0x7FFFu + ((a.u >> 16) & 1u)) >> 16;   // round-to-nearest-even
  union { unsigned short s; bf16 b; } o; o.s = (unsigned short)r;
  return o.b;
}

// ---------------- problem constants ----------------
// B=32, LQ=T=1024, D=1024, h=512. Keys padded 1025 -> 1088.
#define NB_ROWS 32768   // B*LQ
#define DD      1024
#define HH      512
#define TT      1024
#define TP      1088    // padded T+1 (multiple of 64 and 32)
#define BZ      32

// ---------------- generic batched WMMA GEMM ----------------
// C[z] = epilogue( scale * A[z] @ op(B[z]) )
//   TRANS_B=1 : C[m][n] = sum_k A[m*lda+k] * B[n*ldb+k]   (X @ W^T)
//   TRANS_B=0 : C[m][n] = sum_k A[m*lda+k] * B[k*ldb+n]   (attn @ V)
// All of M%128, N%64, K%32 are zero by construction -> no guards.
#define BM 128
#define BN 64
#define BK 32
#define LDP 48   // padded LDS row stride (bf16 elems); 48*2=96 bytes, 16B-aligned rows

template<bool TRANS_B, bool OUT_BF16, bool RELU, bool ACCUM, bool BIAS>
__global__ __launch_bounds__(256) void gemm_wmma(
    const bf16* __restrict__ Ag, const bf16* __restrict__ Bg, void* __restrict__ Cg,
    const float* __restrict__ bias,
    int M, int N, int K, int lda, int ldb, int ldc,
    long strideA, long strideB, long strideC, float scale)
{
  __shared__ bf16 As[BM][LDP];
  __shared__ bf16 Bs[BN][LDP];

  const int z = blockIdx.z;
  Ag += (long)z * strideA;
  Bg += (long)z * strideB;

  const int m0   = blockIdx.y * BM;
  const int n0   = blockIdx.x * BN;
  const int tid  = threadIdx.x;
  const int lane = tid & 31;
  const int wave = tid >> 5;
  const int wm   = (wave & 3) * 32;   // wave row offset in block tile
  const int wn   = (wave >> 2) * 32;  // wave col offset in block tile
  const int r    = lane & 15;
  const int hi   = lane >> 4;

  F8 acc[2][2];
#pragma unroll
  for (int f = 0; f < 2; ++f)
#pragma unroll
    for (int g = 0; g < 2; ++g)
#pragma unroll
      for (int j = 0; j < 8; ++j) acc[f][g].e[j] = 0.0f;

  for (int k0 = 0; k0 < K; k0 += BK) {
    // ---- stage A tile 128x32 (512 chunks of 8 bf16; 2 per thread) ----
#pragma unroll
    for (int rep = 0; rep < 2; ++rep) {
      int c   = tid * 2 + rep;
      int row = c >> 2;
      int col = (c & 3) * 8;
      const bf16* src = Ag + (size_t)(m0 + row) * lda + (k0 + col);
      *(uint4*)&As[row][col] = *(const uint4*)src;
    }
    // ---- stage B tile as Bs[n][k] ----
    if (TRANS_B) {
      int row = tid >> 2;          // n within tile (64 rows)
      int col = (tid & 3) * 8;     // k chunk
      const bf16* src = Bg + (size_t)(n0 + row) * ldb + (k0 + col);
      *(uint4*)&Bs[row][col] = *(const uint4*)src;
    } else {
      int kk = tid & 31;
      int n8 = (tid >> 5) * 8;
      const bf16* src = Bg + (size_t)(k0 + kk) * ldb + (n0 + n8);
#pragma unroll
      for (int j = 0; j < 8; ++j) Bs[n8 + j][kk] = src[j];
    }
    __syncthreads();

    // ---- fragments per CDNA5 wave32 WMMA layouts ----
    BF16x16 a[2], b[2];
#pragma unroll
    for (int f = 0; f < 2; ++f) {
#pragma unroll
      for (int i = 0; i < 8; ++i) {
        int kk = (i < 4) ? (hi * 8 + 2 * i) : (16 + hi * 8 + 2 * (i - 4));
        a[f].e[2 * i]     = As[wm + f * 16 + r][kk];
        a[f].e[2 * i + 1] = As[wm + f * 16 + r][kk + 1];
      }
    }
#pragma unroll
    for (int g = 0; g < 2; ++g) {
#pragma unroll
      for (int i = 0; i < 8; ++i) {
        int kk = (i < 4) ? (hi * 8 + 2 * i) : (16 + hi * 8 + 2 * (i - 4));
        b[g].e[2 * i]     = Bs[wn + g * 16 + r][kk];
        b[g].e[2 * i + 1] = Bs[wn + g * 16 + r][kk + 1];
      }
    }
#pragma unroll
    for (int f = 0; f < 2; ++f)
#pragma unroll
      for (int g = 0; g < 2; ++g)
        acc[f][g].v = __builtin_amdgcn_wmma_f32_16x16x32_bf16(
            false, a[f].v, false, b[g].v, (short)0, acc[f][g].v, false, false);
    __syncthreads();
  }

  // ---- epilogue ----
  float* Cf = (float*)Cg + (long)z * strideC;
  bf16*  Cb = (bf16*)Cg  + (long)z * strideC;
#pragma unroll
  for (int f = 0; f < 2; ++f)
#pragma unroll
    for (int g = 0; g < 2; ++g)
#pragma unroll
      for (int i = 0; i < 8; ++i) {
        int row = m0 + wm + f * 16 + hi * 8 + i;
        int col = n0 + wn + g * 16 + r;
        size_t idx = (size_t)row * ldc + col;
        float val = acc[f][g].e[i] * scale;
        if constexpr (ACCUM) val += Cf[idx];
        if constexpr (BIAS)  val += bias[col];
        if constexpr (RELU)  val = fmaxf(val, 0.0f);
        if constexpr (OUT_BF16) Cb[idx] = to_bf16(val);
        else                    Cf[idx] = val;
      }
}

// ---------------- helper kernels ----------------
__global__ void build_pos(const int* __restrict__ cls, int* __restrict__ pos,
                          int* __restrict__ mask, int T) {
  if (threadIdx.x == 0 && blockIdx.x == 0) {
    int c = 0;
    for (int t = 0; t < T; ++t) {
      int m = (cls[t] != 100) ? 1 : 0;
      mask[t] = m;
      c += m;
      int p = c - 1;
      pos[t] = p < 0 ? 0 : p;
    }
  }
}

__global__ void f32_to_bf16_k(const float* __restrict__ x, bf16* __restrict__ y, long n) {
  long i = (long)blockIdx.x * blockDim.x + threadIdx.x;
  long stride = (long)gridDim.x * blockDim.x;
  for (; i < n; i += stride) y[i] = to_bf16(x[i]);
}

// kf/vf/tf builders: padded key rows (1025..1087 zero), row T = dummy/0/0.
__global__ void build_kvf(const bf16* __restrict__ kp, const float* __restrict__ v,
                          const float* __restrict__ tsp, const bf16* __restrict__ dummy,
                          const int* __restrict__ pos, const int* __restrict__ mask,
                          bf16* __restrict__ kf, bf16* __restrict__ vf, bf16* __restrict__ tf) {
  const int t = blockIdx.x;   // 0..TP-1
  const int b = blockIdx.y;   // 0..BZ-1
  const size_t ro = ((size_t)b * TP + t) * DD;
  const int d0 = threadIdx.x * 4;
  const bf16 zb = to_bf16(0.0f);
  if (t < TT) {
    const int m = mask[t];
    const int p = pos[t];
    const size_t gro = ((size_t)b * TT + p) * DD;
    const size_t tro = ((size_t)b * TT + t) * DD;
#pragma unroll
    for (int j = 0; j < 4; ++j) {
      int d = d0 + j;
      kf[ro + d] = m ? kp[gro + d] : dummy[d];
      vf[ro + d] = m ? to_bf16(v[gro + d]) : zb;
      tf[ro + d] = to_bf16(tsp[tro + d]);
    }
  } else if (t == TT) {
#pragma unroll
    for (int j = 0; j < 4; ++j) {
      int d = d0 + j;
      kf[ro + d] = dummy[d];
      vf[ro + d] = zb;
      tf[ro + d] = zb;
    }
  } else {
#pragma unroll
    for (int j = 0; j < 4; ++j) {
      int d = d0 + j;
      kf[ro + d] = zb;
      vf[ro + d] = zb;
      tf[ro + d] = zb;
    }
  }
}

// row softmax over `valid` cols, zero the pad cols, write bf16 probabilities
__global__ void softmax_rows(const float* __restrict__ S, bf16* __restrict__ P,
                             int valid, int ld) {
  const int rI = blockIdx.x;
  const int tid = threadIdx.x;
  const float* s = S + (size_t)rI * ld;
  bf16* p = P + (size_t)rI * ld;
  __shared__ float red[256];

  float mx = -3.0e38f;
  for (int c = tid; c < valid; c += 256) mx = fmaxf(mx, s[c]);
  red[tid] = mx; __syncthreads();
  for (int o = 128; o > 0; o >>= 1) {
    if (tid < o) red[tid] = fmaxf(red[tid], red[tid + o]);
    __syncthreads();
  }
  mx = red[0]; __syncthreads();

  float sum = 0.0f;
  for (int c = tid; c < valid; c += 256) sum += __expf(s[c] - mx);
  red[tid] = sum; __syncthreads();
  for (int o = 128; o > 0; o >>= 1) {
    if (tid < o) red[tid] += red[tid + o];
    __syncthreads();
  }
  const float inv = 1.0f / red[0];
  const bf16 zb = to_bf16(0.0f);
  for (int c = tid; c < ld; c += 256)
    p[c] = (c < valid) ? to_bf16(__expf(s[c] - mx) * inv) : zb;
}

// x1 = out * residual ; x2 = residual - out  (bf16 for next GEMMs)
__global__ void make_x1x2(const float* __restrict__ outv, const float* __restrict__ q,
                          bf16* __restrict__ x1, bf16* __restrict__ x2, long n) {
  long i = (long)blockIdx.x * blockDim.x + threadIdx.x;
  long stride = (long)gridDim.x * blockDim.x;
  for (; i < n; i += stride) {
    float o = outv[i], r = q[i];
    x1[i] = to_bf16(o * r);
    x2[i] = to_bf16(r - o);
  }
}

// ---------------- launcher ----------------
extern "C" void kernel_launch(void* const* d_in, const int* in_sizes, int n_in,
                              void* d_out, int out_size, void* d_ws, size_t ws_size,
                              hipStream_t stream) {
  const float* q    = (const float*)d_in[0];
  const float* k    = (const float*)d_in[1];
  const float* v    = (const float*)d_in[2];
  const float* tsp  = (const float*)d_in[3];
  const int*   cls  = (const int*)d_in[4];
  const float* wqk  = (const float*)d_in[5];
  const float* dmy  = (const float*)d_in[6];
  const float* w1   = (const float*)d_in[7];
  const float* b1   = (const float*)d_in[8];
  const float* w2   = (const float*)d_in[9];
  const float* b2   = (const float*)d_in[10];
  const float* w3   = (const float*)d_in[11];
  const float* b3   = (const float*)d_in[12];

  char* wsp = (char*)d_ws;
  auto alloc = [&](size_t bytes) {
    char* p = wsp;
    wsp += (bytes + 255) & ~(size_t)255;
    return p;
  };

  bf16* q_bf   = (bf16*)alloc((size_t)NB_ROWS * DD * 2);
  bf16* k_bf   = (bf16*)alloc((size_t)NB_ROWS * DD * 2);
  bf16* qp_bf  = (bf16*)alloc((size_t)NB_ROWS * DD * 2);
  bf16* kp_bf  = (bf16*)alloc((size_t)NB_ROWS * DD * 2);
  bf16* kf     = (bf16*)alloc((size_t)BZ * TP * DD * 2);
  bf16* vf     = (bf16*)alloc((size_t)BZ * TP * DD * 2);
  bf16* tf     = (bf16*)alloc((size_t)BZ * TP * DD * 2);
  float* scores= (float*)alloc((size_t)NB_ROWS * TP * 4);
  bf16* attn   = (bf16*)alloc((size_t)NB_ROWS * TP * 2);
  bf16* wqk_bf = (bf16*)alloc((size_t)DD * DD * 2);
  bf16* w1_bf  = (bf16*)alloc((size_t)HH * DD * 2);
  bf16* w2_bf  = (bf16*)alloc((size_t)HH * DD * 2);
  bf16* w3_bf  = (bf16*)alloc((size_t)DD * 2 * DD * 2);
  bf16* dmy_bf = (bf16*)alloc((size_t)DD * 2);
  int*  pos    = (int*)alloc(TT * 4);
  int*  mask   = (int*)alloc(TT * 4);

  // buffer reuse (lifetimes are disjoint)
  float* out_f32 = scores;   // after softmax, scores region becomes `out`
  bf16*  x1      = kp_bf;    // kp dead after build_kvf
  bf16*  x2      = k_bf;     // k_bf dead after kp GEMM
  bf16*  o12     = qp_bf;    // qp dead after scores GEMM

  float* outp   = (float*)d_out;                         // (NB, D)
  float* tspout = (float*)d_out + (size_t)NB_ROWS * DD;  // (NB, D)

  build_pos<<<1, 1, 0, stream>>>(cls, pos, mask, TT);

  f32_to_bf16_k<<<2048, 256, 0, stream>>>(q, q_bf, (long)NB_ROWS * DD);
  f32_to_bf16_k<<<2048, 256, 0, stream>>>(k, k_bf, (long)NB_ROWS * DD);
  f32_to_bf16_k<<<64, 256, 0, stream>>>(wqk, wqk_bf, (long)DD * DD);
  f32_to_bf16_k<<<32, 256, 0, stream>>>(w1, w1_bf, (long)HH * DD);
  f32_to_bf16_k<<<32, 256, 0, stream>>>(w2, w2_bf, (long)HH * DD);
  f32_to_bf16_k<<<128, 256, 0, stream>>>(w3, w3_bf, (long)DD * 2 * DD);
  f32_to_bf16_k<<<4, 256, 0, stream>>>(dmy, dmy_bf, (long)DD);

  // qp = q @ wqk^T ; kp = k @ wqk^T   (bf16 out, f32 accum)
  gemm_wmma<true, true, false, false, false>
      <<<dim3(DD / BN, NB_ROWS / BM, 1), 256, 0, stream>>>(
      q_bf, wqk_bf, qp_bf, nullptr, NB_ROWS, DD, DD, DD, DD, DD, 0, 0, 0, 1.0f);
  gemm_wmma<true, true, false, false, false>
      <<<dim3(DD / BN, NB_ROWS / BM, 1), 256, 0, stream>>>(
      k_bf, wqk_bf, kp_bf, nullptr, NB_ROWS, DD, DD, DD, DD, DD, 0, 0, 0, 1.0f);

  // gather/mask -> kf, vf, tf (padded to TP rows)
  build_kvf<<<dim3(TP, BZ), 256, 0, stream>>>(kp_bf, v, tsp, dmy_bf, pos, mask, kf, vf, tf);

  // scores = qp @ kf^T / 32   (batched over B, f32 out)
  gemm_wmma<true, false, false, false, false>
      <<<dim3(TP / BN, TT / BM, BZ), 256, 0, stream>>>(
      qp_bf, kf, scores, nullptr, TT, TP, DD, DD, DD, TP,
      (long)TT * DD, (long)TP * DD, (long)TT * TP, 1.0f / 32.0f);

  softmax_rows<<<NB_ROWS, 256, 0, stream>>>(scores, attn, TT + 1, TP);

  // out = attn @ vf (NN, f32 out -> reuses scores buffer)
  gemm_wmma<false, false, false, false, false>
      <<<dim3(DD / BN, TT / BM, BZ), 256, 0, stream>>>(
      attn, vf, out_f32, nullptr, TT, DD, TP, TP, DD, DD,
      (long)TT * TP, (long)TP * DD, (long)TT * DD, 1.0f);

  // tsp_out = attn @ tf  -> second half of d_out (f32)
  gemm_wmma<false, false, false, false, false>
      <<<dim3(DD / BN, TT / BM, BZ), 256, 0, stream>>>(
      attn, tf, tspout, nullptr, TT, DD, TP, TP, DD, DD,
      (long)TT * TP, (long)TP * DD, (long)TT * DD, 1.0f);

  // x1 = out * q ; x2 = q - out
  make_x1x2<<<2048, 256, 0, stream>>>(out_f32, q, x1, x2, (long)NB_ROWS * DD);

  // o1 = relu(x1 @ w1^T + b1) -> o12[:, 0:512] ; o2 likewise -> o12[:, 512:1024]
  gemm_wmma<true, true, true, false, true>
      <<<dim3(HH / BN, NB_ROWS / BM, 1), 256, 0, stream>>>(
      x1, w1_bf, o12, b1, NB_ROWS, HH, DD, DD, DD, DD, 0, 0, 0, 1.0f);
  gemm_wmma<true, true, true, false, true>
      <<<dim3(HH / BN, NB_ROWS / BM, 1), 256, 0, stream>>>(
      x2, w2_bf, o12 + HH, b2, NB_ROWS, HH, DD, DD, DD, DD, 0, 0, 0, 1.0f);

  // output = o12 @ w3[:, :1024]^T + b3  ... then += q @ w3[:, 1024:]^T
  gemm_wmma<true, false, false, false, true>
      <<<dim3(DD / BN, NB_ROWS / BM, 1), 256, 0, stream>>>(
      o12, w3_bf, outp, b3, NB_ROWS, DD, DD, DD, 2 * DD, DD, 0, 0, 0, 1.0f);
  gemm_wmma<true, false, false, true, false>
      <<<dim3(DD / BN, NB_ROWS / BM, 1), 256, 0, stream>>>(
      q_bf, w3_bf + DD, outp, nullptr, NB_ROWS, DD, DD, DD, 2 * DD, DD, 0, 0, 0, 1.0f);
}
